// HybridEncoder_36756330119761
// MI455X (gfx1250) — compile-verified, matrix operand order checked by reading, not run on hardware
//
#include <hip/hip_runtime.h>
#include <hip/hip_bf16.h>

// ---------------------------------------------------------------- types
typedef _Float16 f16;
typedef __attribute__((ext_vector_type(16))) _Float16 v16h;
typedef __attribute__((ext_vector_type(8)))  float    v8f;

union Frag16 { v16h v; unsigned u[8]; };

// Problem constants (match reference)
#define BB   256
#define SS   256
#define DD   256
#define HIDc 512
#define HDc  256
#define GHc  128
#define KNN  5

// K offset within a 32-wide K chunk for f16 A/B fragments (ISA 7.12.2):
// lanes 0-15 hold K-octets {0..7,16..23}, lanes 16-31 hold {8..15,24..31},
// packed 2 f16 per VGPR.
__device__ __forceinline__ int kbase(int j, int half) {
  return (j < 4) ? (half * 8 + 2 * j) : (16 + half * 8 + 2 * (j - 4));
}

// ---------------------------------------------------------------- generic NT GEMM
// C[b] (MxN) = A[b] (MxK row-major f16) * B[b]^T (B stored NxK row-major f16)
// + bias, optional ReLU. 64x64 macro-tile per wave: 4 A-frags x 4 B-frags ->
// 16 v_wmma per 32-wide K chunk (~1 b128 load per WMMA).
#define GEMM_WAVES 8
__global__ __launch_bounds__(256) void gemm_nt_f16(
    const f16* __restrict__ A, const f16* __restrict__ B,
    const float* __restrict__ bias,
    float* __restrict__ C, f16* __restrict__ C16,
    int M, int N, int K,
    long long sA, long long sB, long long sC,
    int relu)
{
  const int lane = threadIdx.x & 31;
  const int wave = threadIdx.x >> 5;
  const int half = lane >> 4;
  const int lrow = lane & 15;

  const int tilesN = N >> 6;                       // 64-wide macro tiles
  const long long tilesTotal = (long long)(M >> 6) * tilesN;
  const long long tile = (long long)blockIdx.x * GEMM_WAVES + wave;
  if (tile >= tilesTotal) return;                  // wave-uniform exit
  const int tm = (int)(tile / tilesN);
  const int tn = (int)(tile % tilesN);

  const f16* Ab = A + (long long)blockIdx.y * sA + (long long)(tm * 64 + lrow) * K;
  const f16* Bb = B + (long long)blockIdx.y * sB + (long long)(tn * 64 + lrow) * K;
  const long long rstride = (long long)16 * K;     // sub-tile row stride

  v8f acc[4][4] = {};
  for (int k0 = 0; k0 < K; k0 += 32) {
    if (k0 + 32 < K) {                             // keep next chunk in near caches
      __builtin_prefetch(Ab + k0 + 32, 0, 3);
      __builtin_prefetch(Bb + k0 + 32, 0, 3);
    }
    Frag16 af[4], bf[4];
#pragma unroll
    for (int i = 0; i < 4; ++i) {
#pragma unroll
      for (int j = 0; j < 8; ++j) {
        const int kk = k0 + kbase(j, half);
        af[i].u[j] = *(const unsigned*)(Ab + i * rstride + kk);
        bf[i].u[j] = *(const unsigned*)(Bb + i * rstride + kk);
      }
    }
#pragma unroll
    for (int i = 0; i < 4; ++i)
#pragma unroll
      for (int j2 = 0; j2 < 4; ++j2)
        acc[i][j2] = __builtin_amdgcn_wmma_f32_16x16x32_f16(
            false, af[i].v, false, bf[j2].v, (short)0, acc[i][j2], false, false);
  }

#pragma unroll
  for (int j2 = 0; j2 < 4; ++j2) {
    const int col = tn * 64 + j2 * 16 + lrow;
    const float bv = bias ? bias[col] : 0.0f;
#pragma unroll
    for (int i = 0; i < 4; ++i) {
#pragma unroll
      for (int r = 0; r < 8; ++r) {
        const int row = tm * 64 + i * 16 + r + half * 8;  // C: VGPR r -> M=r / r+8
        float v = acc[i][j2][r] + bv;
        if (relu) v = fmaxf(v, 0.0f);
        const long long off = (long long)blockIdx.y * sC + (long long)row * N + col;
        if (C)   C[off]   = v;
        if (C16) C16[off] = (f16)v;
      }
    }
  }
}

// ---------------------------------------------------------------- BiGRU recurrence
// One workgroup per direction; h (256x256) lives in LDS as f16 (128 KB).
// Per step: gh = h @ Whh^T + bhh via LDS-fed WMMA (64x32 macro-tiles, 4x24 tile
// grid = 3 tiles/wave uniform over 32 waves), then fused gate math consuming
// precomputed gi (f16, global).
__global__ __launch_bounds__(1024) void gru_kernel(
    const f16* __restrict__ gi_f, const f16* __restrict__ gi_b,
    const f16* __restrict__ Whh_f, const f16* __restrict__ Whh_b,
    const float* __restrict__ bhh_f, const float* __restrict__ bhh_b,
    float* __restrict__ gsc,          // 2 * 256*768 f32 scratch
    float* __restrict__ H)            // (B,S,HID) f32
{
  extern __shared__ f16 hsh[];        // 256*256 f16
  const int dir = blockIdx.x;
  const f16*   gi  = dir ? gi_b  : gi_f;
  const f16*   W   = dir ? Whh_b : Whh_f;
  const float* bhh = dir ? bhh_b : bhh_f;
  float* g = gsc + (long long)dir * (BB * 3 * HDc);

  const int tid  = threadIdx.x;
  const int lane = tid & 31;
  const int wave = tid >> 5;
  const int half = lane >> 4;
  const int lrow = lane & 15;

  for (int i = tid; i < BB * HDc; i += 1024) hsh[i] = (f16)0.0f;
  __syncthreads();

  for (int tt = 0; tt < SS; ++tt) {
    const int t = dir ? (SS - 1 - tt) : tt;

    // ---- phase 1: gh = h @ Whh^T + bhh  (M=256, N=768, K=256)
    for (int tl = wave; tl < 4 * 24; tl += 32) {    // uniform: 3 macro-tiles/wave
      const int tm = tl / 24, tn = tl % 24;         // 64-row x 32-col macro tile
      const f16* hb = hsh + (tm * 64 + lrow) * HDc;             // LDS A-frags
      const f16* Bb = W + (long long)(tn * 32 + lrow) * HDc;    // global B-frags
      v8f acc[4][2] = {};
      for (int k0 = 0; k0 < HDc; k0 += 32) {
        Frag16 af[4], bf[2];
#pragma unroll
        for (int i = 0; i < 4; ++i)
#pragma unroll
          for (int j = 0; j < 8; ++j)
            af[i].u[j] = *(const unsigned*)(hb + i * 16 * HDc + k0 + kbase(j, half));
#pragma unroll
        for (int i = 0; i < 2; ++i)
#pragma unroll
          for (int j = 0; j < 8; ++j)
            bf[i].u[j] = *(const unsigned*)(Bb + (long long)i * 16 * HDc + k0 + kbase(j, half));
#pragma unroll
        for (int i = 0; i < 4; ++i)
#pragma unroll
          for (int j2 = 0; j2 < 2; ++j2)
            acc[i][j2] = __builtin_amdgcn_wmma_f32_16x16x32_f16(
                false, af[i].v, false, bf[j2].v, (short)0, acc[i][j2], false, false);
      }
#pragma unroll
      for (int j2 = 0; j2 < 2; ++j2) {
        const int col = tn * 32 + j2 * 16 + lrow;
        const float bv = bhh[col];
#pragma unroll
        for (int i = 0; i < 4; ++i)
#pragma unroll
          for (int r = 0; r < 8; ++r)
            g[(tm * 64 + i * 16 + r + half * 8) * (3 * HDc) + col] = acc[i][j2][r] + bv;
      }
    }
    __threadfence();
    __syncthreads();

    // ---- phase 2: gates + h update (64 elems/thread)
    for (int e = tid; e < BB * HDc; e += 1024) {
      const int bb = e >> 8, j = e & 255;
      const f16* git = gi + ((long long)bb * SS + t) * (3 * HDc);
      const float ir  = (float)git[j];
      const float iz  = (float)git[HDc + j];
      const float in_ = (float)git[2 * HDc + j];
      const float hr = g[bb * (3 * HDc) + j];
      const float hz = g[bb * (3 * HDc) + HDc + j];
      const float hn = g[bb * (3 * HDc) + 2 * HDc + j];
      const float r = 1.0f / (1.0f + __expf(-(ir + hr)));
      const float z = 1.0f / (1.0f + __expf(-(iz + hz)));
      const float n = tanhf(in_ + r * hn);
      const float hprev = (float)hsh[e];
      const float hnew = (1.0f - z) * n + z * hprev;
      hsh[e] = (f16)hnew;
      H[((long long)bb * SS + t) * HIDc + dir * HDc + j] = hnew;
    }
    __syncthreads();
  }
}

// ---------------------------------------------------------------- helpers
__global__ void cvt_f32_f16(const float* __restrict__ in, f16* __restrict__ out,
                            long long n) {
  const long long e = (long long)blockIdx.x * blockDim.x + threadIdx.x;
  if (e < n) out[e] = (f16)in[e];
}

template <typename T>
__global__ void transpose_to_f16(const T* __restrict__ in, f16* __restrict__ out,
                                 int rows, int cols, long long sIn, long long sOut) {
  const long long e = (long long)blockIdx.x * blockDim.x + threadIdx.x;
  if (e >= (long long)rows * cols) return;
  const int r = (int)(e / cols), c = (int)(e % cols);
  out[(long long)blockIdx.y * sOut + (long long)c * rows + r] =
      (f16)in[(long long)blockIdx.y * sIn + e];
}

// Row L2-normalize H (f32, cols=HID) -> Hn (f16). One wave per row.
__global__ __launch_bounds__(256) void rownorm_kernel(const float* __restrict__ H,
                                                      f16* __restrict__ Hn) {
  const int lane = threadIdx.x & 31;
  const long long row = (long long)blockIdx.x * 8 + (threadIdx.x >> 5);
  const float* p = H + row * HIDc;
  float s = 0.0f;
  for (int c = lane; c < HIDc; c += 32) { const float v = p[c]; s += v * v; }
#pragma unroll
  for (int off = 16; off; off >>= 1) s += __shfl_xor(s, off, 32);
  const float scale = 1.0f / fmaxf(sqrtf(s), 1e-12f);
  f16* q = Hn + row * HIDc;
  for (int c = lane; c < HIDc; c += 32) q[c] = (f16)(p[c] * scale);
}

// Per-row top-K of sim, build A row (ones + self loop) as f16, emit deg^-1/2.
__global__ __launch_bounds__(256) void topk_kernel(const float* __restrict__ sim,
                                                   f16* __restrict__ A,
                                                   float* __restrict__ dinv) {
  const long long row = (long long)blockIdx.x * blockDim.x + threadIdx.x; // b*S+i
  const int i = (int)(row & (SS - 1));
  const float* s = sim + row * SS;
  f16* a = A + row * SS;
  unsigned long long mask[4] = {0, 0, 0, 0};
#pragma unroll
  for (int k = 0; k < KNN; ++k) {
    float best = -1e30f; int bi = 0;
    for (int j = 0; j < SS; ++j) {
      if ((mask[j >> 6] >> (j & 63)) & 1ull) continue;
      const float v = s[j];
      if (v > best) { best = v; bi = j; }
    }
    mask[bi >> 6] |= 1ull << (bi & 63);
  }
  float deg = 0.0f;
  for (int j = 0; j < SS; ++j) {
    float v = ((mask[j >> 6] >> (j & 63)) & 1ull) ? 1.0f : 0.0f;
    if (j == i) v += 1.0f;                         // + eye (self may double)
    a[j] = (f16)v;
    deg += v;
  }
  dinv[row] = rsqrtf(deg + 1e-8f);
}

// A[b,i,j] *= dinv[b,i]*dinv[b,j]  (in-place on f16 adjacency)
__global__ void norm_adj_kernel(f16* __restrict__ A, const float* __restrict__ dinv) {
  const long long e = (long long)blockIdx.x * blockDim.x + threadIdx.x;
  const long long row = e >> 8;                    // b*S + i
  const int j = (int)(e & (SS - 1));
  const long long b = row >> 8;
  A[e] = (f16)((float)A[e] * dinv[row] * dinv[b * SS + j]);
}

// ---------------------------------------------------------------- launcher
extern "C" void kernel_launch(void* const* d_in, const int* in_sizes, int n_in,
                              void* d_out, int out_size, void* d_ws, size_t ws_size,
                              hipStream_t stream) {
  (void)in_sizes; (void)n_in; (void)out_size; (void)ws_size;
  const float* C_masked = (const float*)d_in[0];
  const float* W_proj   = (const float*)d_in[1];
  const float* b_proj   = (const float*)d_in[2];
  const float* Wih_f    = (const float*)d_in[3];
  const float* Whh_f    = (const float*)d_in[4];
  const float* bih_f    = (const float*)d_in[5];
  const float* bhh_f    = (const float*)d_in[6];
  const float* Wih_b    = (const float*)d_in[7];
  const float* Whh_b    = (const float*)d_in[8];
  const float* bih_b    = (const float*)d_in[9];
  const float* bhh_b    = (const float*)d_in[10];
  const float* W_gcn1   = (const float*)d_in[11];
  const float* W_gcn2   = (const float*)d_in[12];
  float* Z_out = (float*)d_out;

  char* ws = (char*)d_ws;
  size_t off = 0;
  auto alloc = [&](size_t bytes) -> void* {
    off = (off + 255) & ~(size_t)255;
    void* p = ws + off; off += bytes; return p;
  };

  const long long nC   = (long long)BB * SS * DD;          // 16.78M
  const long long nCP  = (long long)BB * SS * HIDc;        // 33.55M
  const long long nGI  = (long long)BB * SS * 3 * HDc;     // 50.33M
  const long long nSim = (long long)BB * SS * SS;          // 16.78M
  const long long nX1  = (long long)BB * SS * GHc;         // 8.39M

  f16*   c16     = (f16*)alloc(nC * 2);
  f16*   wproj16 = (f16*)alloc((size_t)HIDc * DD * 2);
  f16*   wihf16  = (f16*)alloc((size_t)3 * HDc * HIDc * 2);
  f16*   whhf16  = (f16*)alloc((size_t)3 * HDc * HDc * 2);
  f16*   wihb16  = (f16*)alloc((size_t)3 * HDc * HIDc * 2);
  f16*   whhb16  = (f16*)alloc((size_t)3 * HDc * HDc * 2);
  f16*   wg1t16  = (f16*)alloc((size_t)GHc * HIDc * 2);
  f16*   wg2t16  = (f16*)alloc((size_t)HIDc * GHc * 2);
  f16*   cproj16 = (f16*)alloc(nCP * 2);
  f16*   gif16   = (f16*)alloc(nGI * 2);
  f16*   gib16   = (f16*)alloc(nGI * 2);
  float* gsc     = (float*)alloc((size_t)2 * BB * 3 * HDc * 4);
  float* Hf      = (float*)alloc(nCP * 4);
  f16*   h16     = (f16*)alloc(nCP * 2);
  f16*   hn16    = (f16*)alloc(nCP * 2);
  float* sim     = (float*)alloc(nSim * 4);
  f16*   a16     = (f16*)alloc(nSim * 2);
  float* dinv    = (float*)alloc((size_t)BB * SS * 4);
  f16*   x116    = (f16*)alloc(nX1 * 2);
  f16*   x1t16   = (f16*)alloc(nX1 * 2);
  f16*   z116    = (f16*)alloc(nX1 * 2);
  f16*   x216    = (f16*)alloc(nCP * 2);
  f16*   x2t16   = (f16*)alloc(nCP * 2);

  // 1) f32 -> f16 conversions
  cvt_f32_f16<<<(unsigned)((nC + 255) / 256), 256, 0, stream>>>(C_masked, c16, nC);
  cvt_f32_f16<<<512, 256, 0, stream>>>(W_proj, wproj16, (long long)HIDc * DD);
  cvt_f32_f16<<<1536, 256, 0, stream>>>(Wih_f, wihf16, (long long)3 * HDc * HIDc);
  cvt_f32_f16<<<768,  256, 0, stream>>>(Whh_f, whhf16, (long long)3 * HDc * HDc);
  cvt_f32_f16<<<1536, 256, 0, stream>>>(Wih_b, wihb16, (long long)3 * HDc * HIDc);
  cvt_f32_f16<<<768,  256, 0, stream>>>(Whh_b, whhb16, (long long)3 * HDc * HDc);
  transpose_to_f16<float><<<dim3(256, 1), 256, 0, stream>>>(W_gcn1, wg1t16, HIDc, GHc, 0, 0);
  transpose_to_f16<float><<<dim3(256, 1), 256, 0, stream>>>(W_gcn2, wg2t16, GHc, HIDc, 0, 0);

  // 2) C_proj = C_seq @ W_proj^T + b_proj   (f16 out)
  // tiles = (65536/64)*(512/64) = 8192 -> 1024 blocks
  gemm_nt_f16<<<1024, 256, 0, stream>>>(c16, wproj16, b_proj, nullptr, cproj16,
                                        BB * SS, HIDc, DD, 0, 0, 0, 0);

  // 3) gi = C_proj @ Wih^T + bih (both directions)
  // tiles = 1024*12 = 12288 -> 1536 blocks
  gemm_nt_f16<<<1536, 256, 0, stream>>>(cproj16, wihf16, bih_f, nullptr, gif16,
                                        BB * SS, 3 * HDc, HIDc, 0, 0, 0, 0);
  gemm_nt_f16<<<1536, 256, 0, stream>>>(cproj16, wihb16, bih_b, nullptr, gib16,
                                        BB * SS, 3 * HDc, HIDc, 0, 0, 0, 0);

  // 4) BiGRU recurrence (h resident in 128 KB LDS)
  gru_kernel<<<2, 1024, BB * HDc * sizeof(f16), stream>>>(
      gif16, gib16, whhf16, whhb16, bhh_f, bhh_b, gsc, Hf);

  // 5) H -> f16 ; row-normalized Hn (f16)
  cvt_f32_f16<<<(unsigned)((nCP + 255) / 256), 256, 0, stream>>>(Hf, h16, nCP);
  rownorm_kernel<<<(BB * SS) / 8, 256, 0, stream>>>(Hf, hn16);

  // 6) sim[b] = Hn[b] @ Hn[b]^T  (batched NT; 16 macro-tiles/batch -> 2 blocks)
  gemm_nt_f16<<<dim3(2, BB), 256, 0, stream>>>(hn16, hn16, nullptr, sim, nullptr,
                                               SS, SS, HIDc,
                                               (long long)SS * HIDc,
                                               (long long)SS * HIDc,
                                               (long long)SS * SS, 0);

  // 7) top-K adjacency + D^{-1/2}, then symmetric normalization (in-place f16)
  topk_kernel<<<(BB * SS) / 256, 256, 0, stream>>>(sim, a16, dinv);
  norm_adj_kernel<<<(unsigned)(nSim / 256), 256, 0, stream>>>(a16, dinv);

  // 8) GCN layer 1: X1 = H @ W_gcn1 ; Z1 = relu(A_norm @ X1)
  // tiles = 1024*2 = 2048 -> 256 blocks
  gemm_nt_f16<<<256, 256, 0, stream>>>(h16, wg1t16, nullptr, nullptr, x116,
                                       BB * SS, GHc, HIDc, 0, 0, 0, 0);
  transpose_to_f16<f16><<<dim3(128, BB), 256, 0, stream>>>(
      x116, x1t16, SS, GHc, (long long)SS * GHc, (long long)SS * GHc);
  // 8 macro-tiles/batch -> 1 block
  gemm_nt_f16<<<dim3(1, BB), 256, 0, stream>>>(a16, x1t16, nullptr, nullptr, z116,
                                               SS, GHc, SS,
                                               (long long)SS * SS,
                                               (long long)GHc * SS,
                                               (long long)SS * GHc, 1);

  // 9) GCN layer 2: X2 = Z1 @ W_gcn2 ; Z = A_norm @ X2  (f32 -> d_out)
  // tiles = 1024*8 = 8192 -> 1024 blocks
  gemm_nt_f16<<<1024, 256, 0, stream>>>(z116, wg2t16, nullptr, nullptr, x216,
                                        BB * SS, HIDc, GHc, 0, 0, 0, 0);
  transpose_to_f16<f16><<<dim3(512, BB), 256, 0, stream>>>(
      x216, x2t16, SS, HIDc, (long long)SS * HIDc, (long long)SS * HIDc);
  // 32 macro-tiles/batch -> 4 blocks
  gemm_nt_f16<<<dim3(4, BB), 256, 0, stream>>>(a16, x2t16, nullptr, Z_out, nullptr,
                                               SS, HIDc, SS,
                                               (long long)SS * SS,
                                               (long long)HIDc * SS,
                                               (long long)SS * HIDc, 0);
}
